// FlashAttention_26164940767817
// MI455X (gfx1250) — matrix operand order repaired and checked by reading, not compiled, not run on hardware
//
#include <hip/hip_runtime.h>
#include <hip/hip_bf16.h>
#include <cstdint>

#define B_ 4
#define N_ 2048
#define C_ 1024
#define H_ 16
#define D_ 64
#define SCALE_ 0.125f   // 1/sqrt(64)

typedef __attribute__((ext_vector_type(8)))  __bf16 v8bf;
typedef __attribute__((ext_vector_type(16))) __bf16 v16bf;
typedef __attribute__((ext_vector_type(8)))  float  v8f;

union Frag16 {
  v16bf v;
  struct { v8bf lo; v8bf hi; } h;
};

__device__ __forceinline__ v8f zero8() {
  v8f z;
#pragma unroll
  for (int i = 0; i < 8; ++i) z[i] = 0.f;
  return z;
}

__device__ __forceinline__ v8f wmma_bf16(v16bf a, v16bf b, v8f c) {
  // (neg_a, A, neg_b, B, c_mod, C, reuse_a, reuse_b)
  return __builtin_amdgcn_wmma_f32_16x16x32_bf16(false, a, false, b, (short)0, c,
                                                 false, false);
}

__device__ __forceinline__ v8bf cvt8(float4 a, float4 b) {
  v8bf r;
  r[0] = (__bf16)a.x; r[1] = (__bf16)a.y; r[2] = (__bf16)a.z; r[3] = (__bf16)a.w;
  r[4] = (__bf16)b.x; r[5] = (__bf16)b.y; r[6] = (__bf16)b.z; r[7] = (__bf16)b.w;
  return r;
}

// ---------------------------------------------------------------------------
// Kernel 0: one-shot f32 -> bf16 conversion (x, qkv_w, out_w). Amortizes the
// conversion once instead of per GEMM tile; keeps GEMM loops cvt-free.
// ---------------------------------------------------------------------------
__global__ __launch_bounds__(256) void cvt_bf16_kernel(
    const float* __restrict__ src, __bf16* __restrict__ dst, int n) {
  const int idx = (blockIdx.x * 256 + threadIdx.x) * 8;
  if (idx + 8 <= n) {
    float4 a = *(const float4*)(src + idx);
    float4 b = *(const float4*)(src + idx + 4);
    *(v8bf*)(dst + idx) = cvt8(a, b);
  }
}

// ---------------------------------------------------------------------------
// Kernel 1: qkv = x @ qkv_w^T + qkv_b   (all-bf16 operands)
//   q,k -> [B,H,N,D] bf16 (WMMA A/B-frag friendly for S = Q K^T)
//   v   -> [B,H,D,N] bf16 (pre-transposed: WMMA B-frag friendly for O = P V)
// Block = 128 threads (4 waves). Block tile 64(M) x 64(N); wave = 16 x 64.
// ---------------------------------------------------------------------------
__global__ __launch_bounds__(128) void qkv_gemm_kernel(
    const __bf16* __restrict__ a, const __bf16* __restrict__ b,
    const float* __restrict__ bias,
    __bf16* __restrict__ qo, __bf16* __restrict__ ko, __bf16* __restrict__ vo) {
  const int wave  = threadIdx.x >> 5;
  const int lane  = threadIdx.x & 31;
  const int l15   = lane & 15;
  const int khalf = lane >> 4;
  const int m0 = blockIdx.x * 64 + wave * 16;
  const int n0 = blockIdx.y * 64;

  v8f acc[4];
#pragma unroll
  for (int j = 0; j < 4; ++j) acc[j] = zero8();

  const __bf16* arow = a + (size_t)(m0 + l15) * C_ + khalf * 8;
  const __bf16* brow[4];
#pragma unroll
  for (int j = 0; j < 4; ++j)
    brow[j] = b + (size_t)(n0 + j * 16 + l15) * C_ + khalf * 8;

#pragma unroll 2
  for (int k0 = 0; k0 < C_; k0 += 32) {
    // batch ALL fragment loads (10 x b128) before any WMMA consumes them
    v8bf alo = *(const v8bf*)(arow + k0);
    v8bf ahi = *(const v8bf*)(arow + k0 + 16);
    v8bf blo[4], bhi[4];
#pragma unroll
    for (int j = 0; j < 4; ++j) {
      blo[j] = *(const v8bf*)(brow[j] + k0);
      bhi[j] = *(const v8bf*)(brow[j] + k0 + 16);
    }
    Frag16 af; af.h.lo = alo; af.h.hi = ahi;
#pragma unroll
    for (int j = 0; j < 4; ++j) {
      Frag16 bf; bf.h.lo = blo[j]; bf.h.hi = bhi[j];
      acc[j] = wmma_bf16(af.v, bf.v, acc[j]);
    }
  }

#pragma unroll
  for (int j = 0; j < 4; ++j) {
    const int col = n0 + j * 16 + l15;
    const float bv = bias[col];
    const int which = col >> 10;      // 0=q 1=k 2=v
    const int cc = col & 1023;
    const int hh = cc >> 6;
    const int dd = cc & 63;
    if (which == 2) {
      // V stored transposed: [B,H,D,N]
#pragma unroll
      for (int i = 0; i < 8; ++i) {
        const int row = m0 + khalf * 8 + i;
        const int bb = row >> 11;
        const int nn = row & 2047;
        vo[(((size_t)(bb * H_ + hh) * D_) + dd) * N_ + nn] =
            (__bf16)(acc[j][i] + bv);
      }
    } else {
      __bf16* dst = (which == 0) ? qo : ko;   // [B,H,N,D]
#pragma unroll
      for (int i = 0; i < 8; ++i) {
        const int row = m0 + khalf * 8 + i;
        const int bb = row >> 11;
        const int nn = row & 2047;
        dst[(((size_t)(bb * H_ + hh) * N_) + nn) * D_ + dd] =
            (__bf16)(acc[j][i] + bv);
      }
    }
  }
}

// ---------------------------------------------------------------------------
// Kernel 2: fused attention. One block (8 waves) per (b,h, 16 query rows).
// Pass 1: online row-max / row-sum (no S storage). Pass 2: recompute S,
// write softmaxed attn to HBM (write-only), accumulate O = P @ V via WMMA,
// with V B-fragments loaded directly from the pre-transposed global layout.
// ---------------------------------------------------------------------------
#define PB_S 40   // padded LDS stride for P rows (bf16 elems)

__global__ __launch_bounds__(256) void attn_kernel(
    const __bf16* __restrict__ q, const __bf16* __restrict__ k,
    const __bf16* __restrict__ vt, float* __restrict__ attn,
    __bf16* __restrict__ o) {
  const int bh = blockIdx.x >> 7;        // b*16 + h
  const int qt = blockIdx.x & 127;
  const int bb = bh >> 4;
  const int hh = bh & 15;
  const int q0 = qt * 16;
  const int wave  = threadIdx.x >> 5;
  const int lane  = threadIdx.x & 31;
  const int l15   = lane & 15;
  const int khalf = lane >> 4;

  const __bf16* qb  = q  + (size_t)bh * N_ * D_;
  const __bf16* kb  = k  + (size_t)bh * N_ * D_;
  const __bf16* vtb = vt + (size_t)bh * D_ * N_;   // [D][N]

  __shared__ float wm[8][16];
  __shared__ float wl[8][16];
  // Pb: 8*16*PB_S*2 = 10240 B ; Obuf (32768 B) aliases the same region.
  __shared__ __align__(32) char smem[8 * 16 * 64 * 4];
  __bf16 (*Pb)[16][PB_S] = (__bf16 (*)[16][PB_S])smem;
  float  (*Ob)[16][64]   = (float  (*)[16][64])smem;

  // Q fragments (reused for every key tile): A-layout, K = head dim.
  Frag16 qf[2];
  {
    const __bf16* qrow = qb + (size_t)(q0 + l15) * D_;
#pragma unroll
    for (int c = 0; c < 2; ++c) {
      qf[c].h.lo = *(const v8bf*)(qrow + c * 32 + khalf * 16);
      qf[c].h.hi = *(const v8bf*)(qrow + c * 32 + khalf * 16 + 8);
    }
  }

  float mrun[8], lrun[8];
#pragma unroll
  for (int i = 0; i < 8; ++i) { mrun[i] = -3.0e38f; lrun[i] = 0.f; }

  // ---------------- pass 1: statistics ----------------
  for (int p = 0; p < 8; ++p) {
    const int pp = wave + 8 * p;           // 32-key pair index
    // prefetch next pair's K rows into cache (global_prefetch_b8)
    if (p < 7)
      __builtin_prefetch(kb + (size_t)((pp + 8) * 32 + l15) * D_, 0, 3);
#pragma unroll
    for (int sub = 0; sub < 2; ++sub) {
      const int kt = pp * 2 + sub;
      Frag16 kf[2];
      const __bf16* krow = kb + (size_t)(kt * 16 + l15) * D_;
      kf[0].h.lo = *(const v8bf*)(krow + khalf * 16);
      kf[0].h.hi = *(const v8bf*)(krow + khalf * 16 + 8);
      kf[1].h.lo = *(const v8bf*)(krow + 32 + khalf * 16);
      kf[1].h.hi = *(const v8bf*)(krow + 32 + khalf * 16 + 8);
      v8f acc = zero8();
      acc = wmma_bf16(qf[0].v, kf[0].v, acc);
      acc = wmma_bf16(qf[1].v, kf[1].v, acc);
      float s[8], tmax[8];
#pragma unroll
      for (int i = 0; i < 8; ++i) { s[i] = acc[i] * SCALE_; tmax[i] = s[i]; }
#pragma unroll
      for (int mask = 1; mask < 16; mask <<= 1)
#pragma unroll
        for (int i = 0; i < 8; ++i)
          tmax[i] = fmaxf(tmax[i], __shfl_xor(tmax[i], mask, 32));
      float psum[8];
#pragma unroll
      for (int i = 0; i < 8; ++i) {
        const float mnew = fmaxf(mrun[i], tmax[i]);
        psum[i] = __expf(s[i] - mnew);
        lrun[i] = lrun[i] * __expf(mrun[i] - mnew);
        mrun[i] = mnew;
      }
#pragma unroll
      for (int mask = 1; mask < 16; mask <<= 1)
#pragma unroll
        for (int i = 0; i < 8; ++i)
          psum[i] += __shfl_xor(psum[i], mask, 32);
#pragma unroll
      for (int i = 0; i < 8; ++i) lrun[i] += psum[i];
    }
  }

  // merge per-wave stats across the 8 waves
  if (l15 == 0) {
#pragma unroll
    for (int i = 0; i < 8; ++i) {
      wm[wave][khalf * 8 + i] = mrun[i];
      wl[wave][khalf * 8 + i] = lrun[i];
    }
  }
  __syncthreads();
  float Mrow[8], invL[8];
#pragma unroll
  for (int i = 0; i < 8; ++i) {
    const int r = khalf * 8 + i;
    float M = -3.0e38f;
    for (int w2 = 0; w2 < 8; ++w2) M = fmaxf(M, wm[w2][r]);
    float L = 0.f;
    for (int w2 = 0; w2 < 8; ++w2) L += wl[w2][r] * __expf(wm[w2][r] - M);
    Mrow[i] = M;
    invL[i] = 1.f / L;
  }

  // ---------------- pass 2: attn write + O accumulate ----------------
  v8f oacc[4];
#pragma unroll
  for (int j = 0; j < 4; ++j) oacc[j] = zero8();
  float* attn_base = attn + ((size_t)bh * N_ + q0) * N_;

  for (int p = 0; p < 8; ++p) {
    const int pp = wave + 8 * p;
    const int kk0 = pp * 32;
    // prefetch this pair's V^T columns and next pair's K rows
    __builtin_prefetch(vtb + (size_t)l15 * N_ + kk0, 0, 3);
    if (p < 7)
      __builtin_prefetch(kb + (size_t)((pp + 8) * 32 + l15) * D_, 0, 3);
#pragma unroll
    for (int sub = 0; sub < 2; ++sub) {
      const int kt = pp * 2 + sub;
      Frag16 kf[2];
      const __bf16* krow = kb + (size_t)(kt * 16 + l15) * D_;
      kf[0].h.lo = *(const v8bf*)(krow + khalf * 16);
      kf[0].h.hi = *(const v8bf*)(krow + khalf * 16 + 8);
      kf[1].h.lo = *(const v8bf*)(krow + 32 + khalf * 16);
      kf[1].h.hi = *(const v8bf*)(krow + 32 + khalf * 16 + 8);
      v8f acc = zero8();
      acc = wmma_bf16(qf[0].v, kf[0].v, acc);
      acc = wmma_bf16(qf[1].v, kf[1].v, acc);
      float pv[8];
#pragma unroll
      for (int i = 0; i < 8; ++i)
        pv[i] = __expf(acc[i] * SCALE_ - Mrow[i]) * invL[i];
      // write softmaxed attention weights (write-only HBM traffic)
#pragma unroll
      for (int i = 0; i < 8; ++i)
        attn_base[(size_t)(khalf * 8 + i) * N_ + kt * 16 + l15] = pv[i];
      // stash P in bf16 A-layout staging (C-layout -> A-layout via LDS)
#pragma unroll
      for (int i = 0; i < 8; ++i)
        Pb[wave][khalf * 8 + i][sub * 16 + l15] = (__bf16)pv[i];
    }
    // O += P(16x32) @ V(32x64): A-frag from Pb, B-frags direct from global V^T
    Frag16 af;
    af.h.lo = *(const v8bf*)&Pb[wave][l15][khalf * 16];
    af.h.hi = *(const v8bf*)&Pb[wave][l15][khalf * 16 + 8];
#pragma unroll
    for (int j = 0; j < 4; ++j) {
      // lane holds column n = d = j*16+l15; K = keys kk0 + khalf*16 + 0..15
      const __bf16* vrow = vtb + (size_t)(j * 16 + l15) * N_ + kk0 + khalf * 16;
      Frag16 bf;
      bf.h.lo = *(const v8bf*)(vrow);
      bf.h.hi = *(const v8bf*)(vrow + 8);
      oacc[j] = wmma_bf16(af.v, bf.v, oacc[j]);
    }
  }

  __syncthreads();   // all waves done with Pb before aliasing as Ob
#pragma unroll
  for (int j = 0; j < 4; ++j)
#pragma unroll
    for (int i = 0; i < 8; ++i)
      Ob[wave][khalf * 8 + i][j * 16 + l15] = oacc[j][i];
  __syncthreads();
  // 8-way reduce, write O as [B,N,C] bf16
  {
    const int e0 = threadIdx.x * 4;
#pragma unroll
    for (int t = 0; t < 4; ++t) {
      const int e = e0 + t;
      const int m = e >> 6, dd = e & 63;
      float s = 0.f;
      for (int w2 = 0; w2 < 8; ++w2) s += Ob[w2][m][dd];
      o[((size_t)(bb * N_ + q0 + m)) * C_ + hh * D_ + dd] = (__bf16)s;
    }
  }
}

// ---------------------------------------------------------------------------
// Kernel 3: out = O @ out_w^T + out_b  (all-bf16 operands, f32 out to d_out)
// ---------------------------------------------------------------------------
__global__ __launch_bounds__(128) void out_gemm_kernel(
    const __bf16* __restrict__ a, const __bf16* __restrict__ b,
    const float* __restrict__ bias, float* __restrict__ out) {
  const int wave  = threadIdx.x >> 5;
  const int lane  = threadIdx.x & 31;
  const int l15   = lane & 15;
  const int khalf = lane >> 4;
  const int m0 = blockIdx.x * 64 + wave * 16;
  const int n0 = blockIdx.y * 64;

  v8f acc[4];
#pragma unroll
  for (int j = 0; j < 4; ++j) acc[j] = zero8();

  const __bf16* arow = a + (size_t)(m0 + l15) * C_ + khalf * 8;
  const __bf16* brow[4];
#pragma unroll
  for (int j = 0; j < 4; ++j)
    brow[j] = b + (size_t)(n0 + j * 16 + l15) * C_ + khalf * 8;

#pragma unroll 2
  for (int k0 = 0; k0 < C_; k0 += 32) {
    v8bf alo = *(const v8bf*)(arow + k0);
    v8bf ahi = *(const v8bf*)(arow + k0 + 16);
    v8bf blo[4], bhi[4];
#pragma unroll
    for (int j = 0; j < 4; ++j) {
      blo[j] = *(const v8bf*)(brow[j] + k0);
      bhi[j] = *(const v8bf*)(brow[j] + k0 + 16);
    }
    Frag16 af; af.h.lo = alo; af.h.hi = ahi;
#pragma unroll
    for (int j = 0; j < 4; ++j) {
      Frag16 bf; bf.h.lo = blo[j]; bf.h.hi = bhi[j];
      acc[j] = wmma_bf16(af.v, bf.v, acc[j]);
    }
  }

#pragma unroll
  for (int j = 0; j < 4; ++j) {
    const int col = n0 + j * 16 + l15;
    const float bv = bias[col];
#pragma unroll
    for (int i = 0; i < 8; ++i) {
      const int row = m0 + khalf * 8 + i;
      out[(size_t)row * C_ + col] = acc[j][i] + bv;
    }
  }
}

// ---------------------------------------------------------------------------
extern "C" void kernel_launch(void* const* d_in, const int* in_sizes, int n_in,
                              void* d_out, int out_size, void* d_ws, size_t ws_size,
                              hipStream_t stream) {
  (void)in_sizes; (void)n_in; (void)out_size; (void)ws_size;
  const float* x     = (const float*)d_in[0];
  const float* qkv_w = (const float*)d_in[1];
  const float* qkv_b = (const float*)d_in[2];
  const float* out_w = (const float*)d_in[3];
  const float* out_b = (const float*)d_in[4];

  float* out  = (float*)d_out;
  float* attn = out + (size_t)B_ * N_ * C_;

  const size_t per = (size_t)B_ * H_ * N_ * D_;   // 8,388,608 elems
  __bf16* qws  = (__bf16*)d_ws;          // [B,H,N,D]
  __bf16* kws  = qws + per;              // [B,H,N,D]
  __bf16* vws  = kws + per;              // [B,H,D,N]  (pre-transposed)
  __bf16* ows  = vws + per;              // [B,N,C]
  __bf16* xbf  = ows + per;              // [B*N, C]        bf16 copy of x
  __bf16* wqkv = xbf + per;              // [3C, C]         bf16 copy of qkv_w
  __bf16* wout = wqkv + (size_t)3 * C_ * C_;   // [C, C]    bf16 copy of out_w

  // 0) one-shot f32 -> bf16 conversions (keeps GEMM hot loops cvt-free)
  {
    const int nx = B_ * N_ * C_;            // 8,388,608
    const int nw1 = 3 * C_ * C_;            // 3,145,728
    const int nw2 = C_ * C_;                // 1,048,576
    cvt_bf16_kernel<<<nx  / (256 * 8), 256, 0, stream>>>(x, xbf, nx);
    cvt_bf16_kernel<<<nw1 / (256 * 8), 256, 0, stream>>>(qkv_w, wqkv, nw1);
    cvt_bf16_kernel<<<nw2 / (256 * 8), 256, 0, stream>>>(out_w, wout, nw2);
  }
  // 1) qkv projection  (M=8192, N=3072, K=1024)
  qkv_gemm_kernel<<<dim3(128, 48), 128, 0, stream>>>(xbf, wqkv, qkv_b,
                                                     qws, kws, vws);
  // 2) fused attention: softmax weights to d_out (write-only) + O accumulate
  attn_kernel<<<dim3(B_ * H_ * (N_ / 16)), 256, 0, stream>>>(qws, kws, vws, attn, ows);
  // 3) output projection (M=8192, N=1024, K=1024)
  out_gemm_kernel<<<dim3(128, 16), 128, 0, stream>>>(ows, wout, out_b, out);
}
